// InteractionGNN_71519795413844
// MI455X (gfx1250) — compile-verified, matrix operand order checked by reading, not run on hardware
//
#include <hip/hip_runtime.h>
#include <hip/hip_bf16.h>

typedef __attribute__((ext_vector_type(16))) __bf16    v16bf;
typedef __attribute__((ext_vector_type(2)))  __bf16    v2bf;
typedef __attribute__((ext_vector_type(8)))  float     v8f;
typedef __attribute__((ext_vector_type(4)))  float     v4f;
typedef __attribute__((ext_vector_type(4)))  unsigned  v4u;

union FragB {
    v16bf v;
    v4u   q[2];
    unsigned u[8];
};

// Native f32 -> bf16 (RNE); compiler lowers fptrunc to the hardware cvt.
__device__ __forceinline__ unsigned short f2bf_s(float x) {
    return __builtin_bit_cast(unsigned short, (__bf16)x);
}
__device__ __forceinline__ unsigned pack_bf16(float a, float b) {
    v2bf t = { (__bf16)a, (__bf16)b };
    return __builtin_bit_cast(unsigned, t);
}

// A-fragment (16x32 bf16) load, per ISA layout:
// lane m = lane&15 owns row m; half = lane>>4 selects K-halves.
// For this lane the fragment is two contiguous 16B chunks of the row:
//   bytes [half*16 .. half*16+15]  (K = half*8 .. half*8+7)
//   bytes [32+half*16 .. +15]      (K = 16+half*8 .. 16+half*8+7)
__device__ __forceinline__ v16bf load_a_bf16(const unsigned short* row, int colBase, int half) {
    FragB f;
    const char* p = (const char*)(row + colBase) + half * 16;
    f.q[0] = *(const v4u*)(p);
    f.q[1] = *(const v4u*)(p + 32);
    return f.v;
}

// A-fragment from an f32 source: the lane's elements are two contiguous
// 8-float runs -> four b128 loads, then pack to bf16. hasHi=false zero-pads
// K in [16,32) of the window (node features, K=16).
__device__ __forceinline__ v16bf load_a_f32cvt(const float* row, int colBase, int half, bool hasHi) {
    FragB f;
    const float* p0 = row + colBase + half * 8;
    v4f c0 = *(const v4f*)(p0);
    v4f c1 = *(const v4f*)(p0 + 4);
    f.u[0] = pack_bf16(c0[0], c0[1]);
    f.u[1] = pack_bf16(c0[2], c0[3]);
    f.u[2] = pack_bf16(c1[0], c1[1]);
    f.u[3] = pack_bf16(c1[2], c1[3]);
    if (hasHi) {
        const float* p1 = row + colBase + 16 + half * 8;
        v4f d0 = *(const v4f*)(p1);
        v4f d1 = *(const v4f*)(p1 + 4);
        f.u[4] = pack_bf16(d0[0], d0[1]);
        f.u[5] = pack_bf16(d0[2], d0[3]);
        f.u[6] = pack_bf16(d1[0], d1[1]);
        f.u[7] = pack_bf16(d1[2], d1[3]);
    } else {
        f.u[4] = f.u[5] = f.u[6] = f.u[7] = 0u;
    }
    return f.v;
}

__device__ __forceinline__ v16bf load_b_frag(const unsigned* __restrict__ Wsw,
                                             int kkb, int nt, int lane) {
    FragB b;
    const unsigned* bp = Wsw + (((size_t)kkb * 4 + nt) * 32 + lane) * 8;
    b.q[0] = *(const v4u*)(bp);
    b.q[1] = *(const v4u*)(bp + 4);
    return b.v;
}

// ---------------------------------------------------------------------------
// Weight swizzle: f32 (Kact x 64) row-major -> bf16 fragment-order layout so a
// B fragment is one contiguous 32B per lane. dst dword index enumerates
// (kkb, nt, lane, v).  K layout mirrors the A-matrix ISA table.
// ---------------------------------------------------------------------------
__global__ void k_swizzle_w(const float* __restrict__ W, unsigned* __restrict__ dst,
                            int Kact, int Kpad) {
    int t = blockIdx.x * blockDim.x + threadIdx.x;
    int total = (Kpad / 32) * 4 * 32 * 8;
    if (t >= total) return;
    int v    = t & 7;
    int lane = (t >> 3) & 31;
    int nt   = (t >> 8) & 3;
    int kkb  = t >> 10;
    int n    = (lane & 15) + nt * 16;
    int half = lane >> 4;
    int k0   = kkb * 32 + (v & 3) * 2 + (v >> 2) * 16 + half * 8;
    float lo = (k0     < Kact) ? W[(size_t)k0 * 64 + n]       : 0.f;
    float hi = (k0 + 1 < Kact) ? W[(size_t)(k0 + 1) * 64 + n] : 0.f;
    dst[t] = pack_bf16(lo, hi);
}

// ---------------------------------------------------------------------------
// Node encoder: h = relu(nodes @ W_ne + b_ne); K=16 padded to 32.
// ---------------------------------------------------------------------------
__global__ void k_node_enc(const float* __restrict__ nodes,
                           const unsigned* __restrict__ Wsw,
                           const float* __restrict__ bias,
                           float* __restrict__ h_f32,
                           unsigned short* __restrict__ h_bf,
                           int nN) {
    int lane = threadIdx.x & 31, wave = threadIdx.x >> 5;
    int tile = blockIdx.x * (blockDim.x >> 5) + wave;
    int nTiles = (nN + 15) >> 4;
    if (tile >= nTiles) return;
    int m = lane & 15, half = lane >> 4;
    bool full = (tile * 16 + 16 <= nN);            // wave-uniform
    int row = tile * 16 + m; if (row > nN - 1) row = nN - 1;

    v16bf A0 = load_a_f32cvt(nodes + (size_t)row * 16, 0, half, false);

    int n0 = lane & 15;
    int rbase = tile * 16 + half * 8;
#pragma unroll
    for (int nt = 0; nt < 4; ++nt) {
        v8f acc = {};
        v16bf B = load_b_frag(Wsw, 0, nt, lane);
        acc = __builtin_amdgcn_wmma_f32_16x16x32_bf16(false, A0, false, B,
                                                      (short)0, acc, false, false);
        int n = n0 + nt * 16;
        float bn = bias[n];
        float*          op = h_f32 + (size_t)rbase * 64 + n;
        unsigned short* bp = h_bf  + (size_t)rbase * 64 + n;
        if (full) {
#pragma unroll
            for (int v = 0; v < 8; ++v) {
                float x = acc[v] + bn;
                x = x > 0.f ? x : 0.f;
                op[v * 64] = x;
                bp[v * 64] = f2bf_s(x);
            }
        } else {
#pragma unroll
            for (int v = 0; v < 8; ++v) {
                if (rbase + v < nN) {
                    float x = acc[v] + bn;
                    x = x > 0.f ? x : 0.f;
                    op[v * 64] = x;
                    bp[v * 64] = f2bf_s(x);
                }
            }
        }
    }
}

// ---------------------------------------------------------------------------
// Edge encoder: e = relu([h[s], h[t]] @ W_ee + b_ee); K=128.
// ---------------------------------------------------------------------------
__global__ void k_edge_enc(const unsigned short* __restrict__ h_bf,
                           const int* __restrict__ si, const int* __restrict__ ei,
                           const unsigned* __restrict__ Wsw,
                           const float* __restrict__ bias,
                           float* __restrict__ e_f32,
                           unsigned short* __restrict__ e_bf,
                           int nE) {
    int lane = threadIdx.x & 31, wave = threadIdx.x >> 5;
    int tile = blockIdx.x * (blockDim.x >> 5) + wave;
    int nTiles = (nE + 15) >> 4;
    if (tile >= nTiles) return;
    int m = lane & 15, half = lane >> 4;
    bool full = (tile * 16 + 16 <= nE);            // wave-uniform
    int edge = tile * 16 + m; if (edge > nE - 1) edge = nE - 1;
    const unsigned short* rowS = h_bf + (size_t)si[edge] * 64;
    const unsigned short* rowT = h_bf + (size_t)ei[edge] * 64;

    v16bf A[4];
    A[0] = load_a_bf16(rowS, 0, half);  A[1] = load_a_bf16(rowS, 32, half);
    A[2] = load_a_bf16(rowT, 0, half);  A[3] = load_a_bf16(rowT, 32, half);

    int n0 = lane & 15;
    int rbase = tile * 16 + half * 8;
#pragma unroll
    for (int nt = 0; nt < 4; ++nt) {
        v8f acc = {};
#pragma unroll
        for (int kkb = 0; kkb < 4; ++kkb) {
            v16bf B = load_b_frag(Wsw, kkb, nt, lane);
            acc = __builtin_amdgcn_wmma_f32_16x16x32_bf16(false, A[kkb], false, B,
                                                          (short)0, acc, false, false);
        }
        int n = n0 + nt * 16;
        float bn = bias[n];
        float*          op = e_f32 + (size_t)rbase * 64 + n;
        unsigned short* bp = e_bf  + (size_t)rbase * 64 + n;
        if (full) {
#pragma unroll
            for (int v = 0; v < 8; ++v) {
                float x = acc[v] + bn;
                x = x > 0.f ? x : 0.f;
                op[v * 64] = x;
                bp[v * 64] = f2bf_s(x);
            }
        } else {
#pragma unroll
            for (int v = 0; v < 8; ++v) {
                if (rbase + v < nE) {
                    float x = acc[v] + bn;
                    x = x > 0.f ? x : 0.f;
                    op[v * 64] = x;
                    bp[v * 64] = f2bf_s(x);
                }
            }
        }
    }
}

// ---------------------------------------------------------------------------
// Scatter-add: msg[ei[e]] += e[e]  (segment_sum). 64 lanes per edge row,
// coalesced reads of e, f32 global atomics into msg.
// ---------------------------------------------------------------------------
__global__ void k_scatter(const float* __restrict__ e, const int* __restrict__ ei,
                          float* __restrict__ msg, int nE) {
    size_t t = (size_t)blockIdx.x * blockDim.x + threadIdx.x;
    if (t >= (size_t)nE * 64) return;
    int edge = (int)(t >> 6);
    int c = (int)(t & 63);
    atomicAdd(&msg[(size_t)ei[edge] * 64 + c], e[t]);
}

// ---------------------------------------------------------------------------
// Edge network: e = relu([h[s], h[t], e] @ W_en + b_en) + e; K=192, in place.
// ---------------------------------------------------------------------------
__global__ void k_edge_net(const unsigned short* __restrict__ h_bf,
                           unsigned short* __restrict__ e_bf,
                           float* __restrict__ e_f32,
                           const int* __restrict__ si, const int* __restrict__ ei,
                           const unsigned* __restrict__ Wsw,
                           const float* __restrict__ bias,
                           int nE) {
    int lane = threadIdx.x & 31, wave = threadIdx.x >> 5;
    int tile = blockIdx.x * (blockDim.x >> 5) + wave;
    int nTiles = (nE + 15) >> 4;
    if (tile >= nTiles) return;
    int m = lane & 15, half = lane >> 4;
    bool full = (tile * 16 + 16 <= nE);            // wave-uniform
    int edge = tile * 16 + m; if (edge > nE - 1) edge = nE - 1;
    const unsigned short* rowS = h_bf + (size_t)si[edge] * 64;
    const unsigned short* rowT = h_bf + (size_t)ei[edge] * 64;
    const unsigned short* rowE = e_bf + (size_t)edge * 64;

    v16bf A[6];
    A[0] = load_a_bf16(rowS, 0, half);  A[1] = load_a_bf16(rowS, 32, half);
    A[2] = load_a_bf16(rowT, 0, half);  A[3] = load_a_bf16(rowT, 32, half);
    A[4] = load_a_bf16(rowE, 0, half);  A[5] = load_a_bf16(rowE, 32, half);

    int n0 = lane & 15;
    int rbase = tile * 16 + half * 8;
#pragma unroll
    for (int nt = 0; nt < 4; ++nt) {
        v8f acc = {};
#pragma unroll
        for (int kkb = 0; kkb < 6; ++kkb) {
            v16bf B = load_b_frag(Wsw, kkb, nt, lane);
            acc = __builtin_amdgcn_wmma_f32_16x16x32_bf16(false, A[kkb], false, B,
                                                          (short)0, acc, false, false);
        }
        int n = n0 + nt * 16;
        float bn = bias[n];
        float*          op = e_f32 + (size_t)rbase * 64 + n;
        unsigned short* bp = e_bf  + (size_t)rbase * 64 + n;
        if (full) {
#pragma unroll
            for (int v = 0; v < 8; ++v) {
                float x = acc[v] + bn;
                x = x > 0.f ? x : 0.f;
                float nv = x + op[v * 64];     // residual
                op[v * 64] = nv;
                bp[v * 64] = f2bf_s(nv);
            }
        } else {
#pragma unroll
            for (int v = 0; v < 8; ++v) {
                if (rbase + v < nE) {
                    float x = acc[v] + bn;
                    x = x > 0.f ? x : 0.f;
                    float nv = x + op[v * 64];
                    op[v * 64] = nv;
                    bp[v * 64] = f2bf_s(nv);
                }
            }
        }
    }
}

// ---------------------------------------------------------------------------
// Node network: h = relu([h, msg] @ W_nn + b_nn) + h; K=128, in place.
// msg is f32 (atomic accumulation target), converted on the fly.
// ---------------------------------------------------------------------------
__global__ void k_node_net(unsigned short* __restrict__ h_bf,
                           float* __restrict__ h_f32,
                           const float* __restrict__ msg,
                           const unsigned* __restrict__ Wsw,
                           const float* __restrict__ bias,
                           int nN) {
    int lane = threadIdx.x & 31, wave = threadIdx.x >> 5;
    int tile = blockIdx.x * (blockDim.x >> 5) + wave;
    int nTiles = (nN + 15) >> 4;
    if (tile >= nTiles) return;
    int m = lane & 15, half = lane >> 4;
    bool full = (tile * 16 + 16 <= nN);            // wave-uniform
    int row = tile * 16 + m; if (row > nN - 1) row = nN - 1;
    const unsigned short* rowH = h_bf + (size_t)row * 64;
    const float*          rowM = msg  + (size_t)row * 64;

    v16bf A[4];
    A[0] = load_a_bf16(rowH, 0, half);         A[1] = load_a_bf16(rowH, 32, half);
    A[2] = load_a_f32cvt(rowM, 0, half, true); A[3] = load_a_f32cvt(rowM, 32, half, true);

    int n0 = lane & 15;
    int rbase = tile * 16 + half * 8;
#pragma unroll
    for (int nt = 0; nt < 4; ++nt) {
        v8f acc = {};
#pragma unroll
        for (int kkb = 0; kkb < 4; ++kkb) {
            v16bf B = load_b_frag(Wsw, kkb, nt, lane);
            acc = __builtin_amdgcn_wmma_f32_16x16x32_bf16(false, A[kkb], false, B,
                                                          (short)0, acc, false, false);
        }
        int n = n0 + nt * 16;
        float bn = bias[n];
        float*          op = h_f32 + (size_t)rbase * 64 + n;
        unsigned short* bp = h_bf  + (size_t)rbase * 64 + n;
        if (full) {
#pragma unroll
            for (int v = 0; v < 8; ++v) {
                float x = acc[v] + bn;
                x = x > 0.f ? x : 0.f;
                float nv = x + op[v * 64];     // residual
                op[v * 64] = nv;
                bp[v * 64] = f2bf_s(nv);
            }
        } else {
#pragma unroll
            for (int v = 0; v < 8; ++v) {
                if (rbase + v < nN) {
                    float x = acc[v] + bn;
                    x = x > 0.f ? x : 0.f;
                    float nv = x + op[v * 64];
                    op[v * 64] = nv;
                    bp[v * 64] = f2bf_s(nv);
                }
            }
        }
    }
}

// ---------------------------------------------------------------------------
// Prediction head: out[e] = [h[s], h[t], e] . W_pred + b_pred.  One wave per
// edge, f32 masters for accuracy, wave32 shuffle reduction.
// ---------------------------------------------------------------------------
__global__ void k_pred(const float* __restrict__ h, const float* __restrict__ e,
                       const int* __restrict__ si, const int* __restrict__ ei,
                       const float* __restrict__ Wp, const float* __restrict__ bp,
                       float* __restrict__ out, int nE) {
    int lane = threadIdx.x & 31, wave = threadIdx.x >> 5;
    int edge = blockIdx.x * (blockDim.x >> 5) + wave;
    if (edge >= nE) return;
    const float* rs = h + (size_t)si[edge] * 64;
    const float* rt = h + (size_t)ei[edge] * 64;
    const float* re = e + (size_t)edge * 64;
    float p = 0.f;
#pragma unroll
    for (int c0 = 0; c0 < 64; c0 += 32) {
        int c = c0 + lane;
        p += rs[c] * Wp[c] + rt[c] * Wp[64 + c] + re[c] * Wp[128 + c];
    }
#pragma unroll
    for (int off = 16; off; off >>= 1) p += __shfl_down(p, off, 32);
    if (lane == 0) out[edge] = p + bp[0];
}

// ---------------------------------------------------------------------------

extern "C" void kernel_launch(void* const* d_in, const int* in_sizes, int n_in,
                              void* d_out, int out_size, void* d_ws, size_t ws_size,
                              hipStream_t stream) {
    const float* nodes  = (const float*)d_in[0];
    const int*   si     = (const int*)d_in[1];
    const int*   ei     = (const int*)d_in[2];
    const float* W_ne   = (const float*)d_in[3];
    const float* b_ne   = (const float*)d_in[4];
    const float* W_ee   = (const float*)d_in[5];
    const float* b_ee   = (const float*)d_in[6];
    const float* W_nn   = (const float*)d_in[7];
    const float* b_nn   = (const float*)d_in[8];
    const float* W_en   = (const float*)d_in[9];
    const float* b_en   = (const float*)d_in[10];
    const float* W_pred = (const float*)d_in[11];
    const float* b_pred = (const float*)d_in[12];

    const int N = in_sizes[0] / 16;   // nodes: (N, 16)
    const int E = in_sizes[1];        // start_index: (E,)

    // --- workspace carve-out (256B aligned slices) ---
    char* ws = (char*)d_ws;
    size_t off = 0;
    auto carve = [&](size_t bytes) -> void* {
        void* p = ws + off;
        off += (bytes + 255) & ~(size_t)255;
        return p;
    };
    float*          e_f32  = (float*)carve((size_t)E * 64 * 4);
    unsigned short* e_bf   = (unsigned short*)carve((size_t)E * 64 * 2);
    float*          h_f32  = (float*)carve((size_t)N * 64 * 4);
    unsigned short* h_bf   = (unsigned short*)carve((size_t)N * 64 * 2);
    float*          msg    = (float*)carve((size_t)N * 64 * 4);
    unsigned*       Wne_sw = (unsigned*)carve(1024 * 4);   // Kpad=32
    unsigned*       Wee_sw = (unsigned*)carve(4096 * 4);   // K=128
    unsigned*       Wnn_sw = (unsigned*)carve(4096 * 4);   // K=128
    unsigned*       Wen_sw = (unsigned*)carve(6144 * 4);   // K=192

    // --- weight swizzle into WMMA B-fragment order (bf16) ---
    k_swizzle_w<<<4,  256, 0, stream>>>(W_ne, Wne_sw, 16, 32);
    k_swizzle_w<<<16, 256, 0, stream>>>(W_ee, Wee_sw, 128, 128);
    k_swizzle_w<<<16, 256, 0, stream>>>(W_nn, Wnn_sw, 128, 128);
    k_swizzle_w<<<24, 256, 0, stream>>>(W_en, Wen_sw, 192, 192);

    const int tilesN = (N + 15) / 16;
    const int tilesE = (E + 15) / 16;
    const int wpb = 8;                 // 8 waves (256 threads) per block

    // --- encoders ---
    k_node_enc<<<(tilesN + wpb - 1) / wpb, 256, 0, stream>>>(
        nodes, Wne_sw, b_ne, h_f32, h_bf, N);
    k_edge_enc<<<(tilesE + wpb - 1) / wpb, 256, 0, stream>>>(
        h_bf, si, ei, Wee_sw, b_ee, e_f32, e_bf, E);

    // --- message-passing iterations ---
    const int scatterBlocks = (int)(((size_t)E * 64 + 255) / 256);
    for (int it = 0; it < 4; ++it) {
        hipMemsetAsync(msg, 0, (size_t)N * 64 * 4, stream);
        k_scatter<<<scatterBlocks, 256, 0, stream>>>(e_f32, ei, msg, E);
        // edge net must read OLD h -> run before node net updates h
        k_edge_net<<<(tilesE + wpb - 1) / wpb, 256, 0, stream>>>(
            h_bf, e_bf, e_f32, si, ei, Wen_sw, b_en, E);
        k_node_net<<<(tilesN + wpb - 1) / wpb, 256, 0, stream>>>(
            h_bf, h_f32, msg, Wnn_sw, b_nn, N);
    }

    // --- prediction head ---
    k_pred<<<(E + wpb - 1) / wpb, 256, 0, stream>>>(
        h_f32, e_f32, si, ei, W_pred, b_pred, (float*)d_out, E);
}
